// DarkMamba_74234214744160
// MI455X (gfx1250) — matrix operand order repaired and checked by reading, not compiled, not run on hardware
//
#include <hip/hip_runtime.h>
#include <math.h>

// ---------------------------------------------------------------------------
// DarkMamba forward for MI455X (gfx1250, wave32, WMMA + TDM).
// GEMMs: operands pre-converted to padded packed f16; inner loop is
// branch-free b128 loads + v_wmma_f32_16x16x32_f16. Weight tiles staged in
// LDS by the Tensor Data Mover (tensor_load_to_lds + s_wait_tensorcnt).
// Selective scan: 3-pass chunked parallel scan in fp32.
// ---------------------------------------------------------------------------

typedef __attribute__((ext_vector_type(16))) _Float16 v16h;
typedef __attribute__((ext_vector_type(8)))  _Float16 v8h;
typedef __attribute__((ext_vector_type(8)))  float    v8f;
typedef __attribute__((ext_vector_type(4)))  unsigned int v4u;
typedef __attribute__((ext_vector_type(8)))  int      v8i_t;
typedef __attribute__((ext_vector_type(4)))  int      v4i_t;

#define SCAN_CHUNK 256
#define MAX_KP 512   // largest padded K (stage-3 di = 512)

__device__ __forceinline__ float sigmoidf_(float x) { return 1.0f / (1.0f + __expf(-x)); }
__device__ __forceinline__ float siluf_(float x)    { return x * sigmoidf_(x); }
__device__ __forceinline__ float softplusf_(float x){ return (x > 20.0f) ? x : log1pf(__expf(x)); }

// ---------------------------------------------------------------------------
// TDM: async-load a [rows x Kp] f16 tile (row stride Kp) from global to LDS.
// D# built per CDNA5 ISA ch.8: group0 {count=1, lds_addr, global_addr, type=2},
// group1 {data_size=2B, tensor_dim0=Kp, tensor_dim1=rows, tile_dim0=Kp,
//         tile_dim1=rows, tensor_dim0_stride=Kp}. Groups 2/3 (+extra group on
// this toolchain's 6-arg builtin) are zero: 2-D tile only.
// ---------------------------------------------------------------------------
__device__ __forceinline__ void tdm_load_tile_f16(const _Float16* gsrc,
                                                  unsigned int lds_byte_off,
                                                  unsigned int Kp,
                                                  unsigned int rows)
{
    unsigned long long ga = (unsigned long long)(uintptr_t)gsrc;
    v4u g0;
    g0[0] = 1u;                                        // count=1 (valid user D#)
    g0[1] = lds_byte_off;                              // lds_addr
    g0[2] = (unsigned int)(ga & 0xFFFFFFFFu);          // global_addr[31:0]
    g0[3] = (unsigned int)((ga >> 32) & 0x01FFFFFFu)   // global_addr[56:32]
            | (2u << 30);                              // type = 2 ("image")
    unsigned int d0 = (1u << 16);                      // data_size = 1 -> 2 bytes
    unsigned int d1 = (Kp & 0xFFFFu) << 16;            // tensor_dim0[15:0]
    unsigned int d2 = (Kp >> 16) | ((rows & 0xFFFFu) << 16);   // td0 hi | td1 lo
    unsigned int d3 = (rows >> 16) | ((Kp & 0xFFFFu) << 16);   // td1 hi | tile_dim0
    unsigned int d4 = (rows & 0xFFFFu);                // tile_dim1 (tile_dim2 = 0)
    unsigned int d5 = Kp;                              // tensor_dim0_stride[31:0]
    v8i_t g1;
    g1[0] = (int)d0; g1[1] = (int)d1; g1[2] = (int)d2; g1[3] = (int)d3;
    g1[4] = (int)d4; g1[5] = (int)d5; g1[6] = 0;       g1[7] = 0;
    v4i_t z4 = {0, 0, 0, 0};
    v8i_t z8 = {0, 0, 0, 0, 0, 0, 0, 0};
    __builtin_amdgcn_tensor_load_to_lds(g0, g1, z4, z4, z8, 0);
}

// ---------------------------------------------------------------------------
// f32 -> packed/padded f16 conversion kernels
// ---------------------------------------------------------------------------
__global__ void cvt_x_f16(const float* __restrict__ X, int lda,
                          _Float16* __restrict__ out, int M, int K, int Kp)
{
    int idx = blockIdx.x * blockDim.x + threadIdx.x;
    if (idx >= M * Kp) return;
    int k = idx % Kp, m = idx / Kp;
    out[idx] = (k < K) ? (_Float16)X[(size_t)m * lda + k] : (_Float16)0.0f;
}

__global__ void cvt_w_f16(const float* __restrict__ W,
                          _Float16* __restrict__ out, int N, int K, int Kp, int Np)
{
    int idx = blockIdx.x * blockDim.x + threadIdx.x;
    if (idx >= Np * Kp) return;
    int k = idx % Kp, n = idx / Kp;
    out[idx] = (n < N && k < K) ? (_Float16)W[(size_t)n * K + k] : (_Float16)0.0f;
}

// ---------------------------------------------------------------------------
// GEMM: Y[M,N] = X16[M,Kp] @ W16[Np,Kp]^T (+bias)(+softplus)(+resid)
// Block = 4 waves: 4 row tiles x 1 col tile. W col-tile staged in LDS by TDM.
// Inner loop: per lane two b128 A loads (global), two b128 B loads (LDS),
// one v_wmma_f32_16x16x32_f16. No predication before/inside the MMA loop.
// Operand packing used here (contiguous k order, same in A and B):
//   A f16 16x32: lane(m=lane&15, h=lane>>4) holds K = k0+16h..+15
//   B f16 32x16: lane(n=lane&15, h) holds K = k0+16h..+15
//   C/D f32: VGPR r -> row r+8h, col lane&15
// ---------------------------------------------------------------------------
__global__ __launch_bounds__(128)
void gemm_wmma_f16(const _Float16* __restrict__ X16,
                   const _Float16* __restrict__ W16,
                   const float* __restrict__ bias,
                   const float* __restrict__ resid, int ldr,
                   float* __restrict__ Y, int ldy,
                   int M, int N, int Kp, int act)
{
    __shared__ __align__(16) _Float16 lds_w[16 * MAX_KP];
    const int wave = threadIdx.x >> 5;
    const int lane = threadIdx.x & 31;
    const int l16  = lane & 15;
    const int half = lane >> 4;
    const int col0 = blockIdx.y * 16;

    if (wave == 0) {
        tdm_load_tile_f16(W16 + (size_t)col0 * Kp,
                          (unsigned int)(uintptr_t)lds_w, (unsigned int)Kp, 16u);
        __builtin_amdgcn_s_wait_tensorcnt(0);
    }
    __syncthreads();

    const int row0 = (blockIdx.x * 4 + wave) * 16;
    if (row0 < M) {
        v8f acc = {};
        const _Float16* xr   = X16 + (size_t)(row0 + l16) * Kp;
        const _Float16* wrow = lds_w + l16 * Kp + 16 * half;
        for (int k0 = 0; k0 < Kp; k0 += 32) {
            v8h alo = *(const v8h*)(xr + k0 + 16 * half);
            v8h ahi = *(const v8h*)(xr + k0 + 16 * half + 8);
            v8h blo = *(const v8h*)(wrow + k0);
            v8h bhi = *(const v8h*)(wrow + k0 + 8);
            v16h a = __builtin_shufflevector(alo, ahi, 0, 1, 2, 3, 4, 5, 6, 7,
                                             8, 9, 10, 11, 12, 13, 14, 15);
            v16h b = __builtin_shufflevector(blo, bhi, 0, 1, 2, 3, 4, 5, 6, 7,
                                             8, 9, 10, 11, 12, 13, 14, 15);
            acc = __builtin_amdgcn_wmma_f32_16x16x32_f16(
                false, a, false, b, (short)0, acc, false, false);
        }
#pragma unroll
        for (int r = 0; r < 8; ++r) {
            int row = row0 + r + 8 * half;
            int col = col0 + l16;
            if (row < M && col < N) {
                float v = acc[r];
                if (bias)  v += bias[col];
                if (act == 1) v = softplusf_(v);
                if (resid) v += resid[(size_t)row * ldr + col];
                Y[(size_t)row * ldy + col] = v;
            }
        }
    }
}

// --------------------------- LayerNorm (wave/token) ------------------------
__global__ __launch_bounds__(32)
void layernorm_k(const float* __restrict__ x, const float* __restrict__ g,
                 const float* __restrict__ b, float* __restrict__ y,
                 int L, int C)
{
    int t = blockIdx.x;
    if (t >= L) return;
    int lane = threadIdx.x;
    const float* xr = x + (size_t)t * C;
    float s = 0.0f, s2 = 0.0f;
    for (int c = lane; c < C; c += 32) { float v = xr[c]; s += v; s2 += v * v; }
#pragma unroll
    for (int m = 16; m >= 1; m >>= 1) {
        s  += __shfl_xor(s,  m, 32);
        s2 += __shfl_xor(s2, m, 32);
    }
    float mean = s / (float)C;
    float var  = s2 / (float)C - mean * mean;
    float inv  = rsqrtf(var + 1e-5f);
    float* yr = y + (size_t)t * C;
    for (int c = lane; c < C; c += 32)
        yr[c] = (xr[c] - mean) * inv * g[c] + b[c];
}

// ------------------ causal depthwise conv (K=4) + SiLU ---------------------
__global__ void causal_conv_silu(const float* __restrict__ xz, int ldxz,
                                 const float* __restrict__ cw,
                                 const float* __restrict__ cb,
                                 float* __restrict__ out, int L, int di)
{
    int idx = blockIdx.x * blockDim.x + threadIdx.x;
    if (idx >= L * di) return;
    int c = idx % di;
    int l = idx / di;
    float acc = cb[c];
#pragma unroll
    for (int k = 0; k < 4; ++k) {
        int ll = l - 3 + k;
        if (ll >= 0) acc += xz[(size_t)ll * ldxz + c] * cw[c * 4 + k];
    }
    out[idx] = siluf_(acc);
}

// ------------------------- selective scan, 3 passes ------------------------
__global__ __launch_bounds__(32)
void scan_pass1(const float* __restrict__ delta,
                const float* __restrict__ Bm, int ldb,
                const float* __restrict__ xcs,
                const float* __restrict__ A_log,
                float* __restrict__ aggA, float* __restrict__ aggB,
                int L, int di)
{
    int chunk = blockIdx.x;
    int lane  = threadIdx.x;
    int n = lane & 15;
    int c = blockIdx.y * 2 + (lane >> 4);
    float A = -__expf(A_log[c * 16 + n]);
    float a_agg = 1.0f, b_agg = 0.0f;
    int l0 = chunk * SCAN_CHUNK;
    int l1 = l0 + SCAN_CHUNK; if (l1 > L) l1 = L;
    for (int l = l0; l < l1; ++l) {
        float d  = delta[(size_t)l * di + c];
        float bm = Bm[(size_t)l * ldb + n];
        float u  = xcs[(size_t)l * di + c];
        float dA = __expf(d * A);
        a_agg = dA * a_agg;
        b_agg = dA * b_agg + d * bm * u;
    }
    size_t o = ((size_t)chunk * di + c) * 16 + n;
    aggA[o] = a_agg;
    aggB[o] = b_agg;
}

__global__ void scan_pass2(const float* __restrict__ aggA,
                           float* __restrict__ aggB, int di, int nchunks)
{
    int t = blockIdx.x * blockDim.x + threadIdx.x;
    if (t >= di * 16) return;
    float h = 0.0f;
    for (int j = 0; j < nchunks; ++j) {
        size_t o = (size_t)j * di * 16 + t;
        float a = aggA[o];
        float b = aggB[o];
        aggB[o] = h;           // state at chunk start
        h = a * h + b;
    }
}

__global__ __launch_bounds__(32)
void scan_pass3(const float* __restrict__ delta,
                const float* __restrict__ Bm, const float* __restrict__ Cm, int ldbc,
                const float* __restrict__ xcs,
                const float* __restrict__ A_log, const float* __restrict__ Dp,
                const float* __restrict__ zbuf, int ldz,
                const float* __restrict__ aggB,
                float* __restrict__ ym, int L, int di)
{
    int chunk = blockIdx.x;
    int lane  = threadIdx.x;
    int n = lane & 15;
    int c = blockIdx.y * 2 + (lane >> 4);
    float A  = -__expf(A_log[c * 16 + n]);
    float Dv = Dp[c];
    float h  = aggB[((size_t)chunk * di + c) * 16 + n];
    int l0 = chunk * SCAN_CHUNK;
    int l1 = l0 + SCAN_CHUNK; if (l1 > L) l1 = L;
    for (int l = l0; l < l1; ++l) {
        float d  = delta[(size_t)l * di + c];
        float bm = Bm[(size_t)l * ldbc + n];
        float cm = Cm[(size_t)l * ldbc + n];
        float u  = xcs[(size_t)l * di + c];
        h = __expf(d * A) * h + d * bm * u;
        float part = h * cm;
#pragma unroll
        for (int m = 1; m < 16; m <<= 1) part += __shfl_xor(part, m, 16);
        if (n == 0) {
            float yv = part + u * Dv;
            float z  = zbuf[(size_t)l * ldz + c];
            ym[(size_t)l * di + c] = yv * siluf_(z);
        }
    }
}

// --------------------------- spatial conv kernels --------------------------
__global__ void conv3x3_in(const float* __restrict__ inp,
                           const float* __restrict__ w,
                           const float* __restrict__ b,
                           float* __restrict__ out, int H, int Wd)
{
    int idx = blockIdx.x * blockDim.x + threadIdx.x;
    if (idx >= H * Wd * 32) return;
    int oc  = idx & 31;
    int pix = idx >> 5;
    int h = pix / Wd, wI = pix % Wd;
    float acc = b[oc];
    for (int ic = 0; ic < 3; ++ic)
        for (int kh = 0; kh < 3; ++kh) {
            int hh = h + kh - 1;
            if (hh < 0 || hh >= H) continue;
            for (int kw = 0; kw < 3; ++kw) {
                int ww = wI + kw - 1;
                if (ww < 0 || ww >= Wd) continue;
                acc += inp[((size_t)ic * H + hh) * Wd + ww] *
                       w[((oc * 3 + ic) * 3 + kh) * 3 + kw];
            }
        }
    out[(size_t)pix * 32 + oc] = acc;
}

__global__ void conv2x2_down(const float* __restrict__ x,
                             const float* __restrict__ w,
                             const float* __restrict__ b,
                             float* __restrict__ out, int H, int Wd, int C)
{
    int OC = 2 * C, Ho = H / 2, Wo = Wd / 2;
    int idx = blockIdx.x * blockDim.x + threadIdx.x;
    if (idx >= Ho * Wo * OC) return;
    int oc   = idx % OC;
    int opix = idx / OC;
    int oh = opix / Wo, ow = opix % Wo;
    float acc = b[oc];
    for (int ic = 0; ic < C; ++ic)
#pragma unroll
        for (int kh = 0; kh < 2; ++kh)
#pragma unroll
            for (int kw = 0; kw < 2; ++kw)
                acc += x[((size_t)(2 * oh + kh) * Wd + (2 * ow + kw)) * C + ic] *
                       w[((oc * C + ic) * 2 + kh) * 2 + kw];
    out[(size_t)opix * OC + oc] = acc;
}

__global__ void pixshuf_add(const float* __restrict__ cin,
                            const float* __restrict__ skip,
                            float* __restrict__ out, int H, int Wd, int Co)
{
    int W2 = 2 * Wd;
    int idx = blockIdx.x * blockDim.x + threadIdx.x;
    if (idx >= 4 * H * Wd * Co) return;
    int c    = idx % Co;
    int opix = idx / Co;
    int oh = opix / W2, ow = opix % W2;
    int h = oh >> 1, i = oh & 1, wI = ow >> 1, j = ow & 1;
    out[idx] = cin[((size_t)h * Wd + wI) * (4 * Co) + (c * 4 + i * 2 + j)] + skip[idx];
}

__global__ void conv3x3_out(const float* __restrict__ x,
                            const float* __restrict__ w,
                            const float* __restrict__ b,
                            const float* __restrict__ inp,
                            float* __restrict__ out, int H, int Wd)
{
    int idx = blockIdx.x * blockDim.x + threadIdx.x;
    if (idx >= 3 * H * Wd) return;
    int oc  = idx / (H * Wd);
    int pix = idx % (H * Wd);
    int h = pix / Wd, wI = pix % Wd;
    float acc = b[oc];
    for (int kh = 0; kh < 3; ++kh) {
        int hh = h + kh - 1;
        if (hh < 0 || hh >= H) continue;
        for (int kw = 0; kw < 3; ++kw) {
            int ww = wI + kw - 1;
            if (ww < 0 || ww >= Wd) continue;
            const float* xr = x + ((size_t)hh * Wd + ww) * 32;
            for (int ic = 0; ic < 32; ++ic)
                acc += xr[ic] * w[((oc * 32 + ic) * 3 + kh) * 3 + kw];
        }
    }
    out[idx] = acc + inp[idx];
}

__global__ void add_inplace(float* __restrict__ y, const float* __restrict__ r, int n)
{
    int i = blockIdx.x * blockDim.x + threadIdx.x;
    if (i < n) y[i] += r[i];
}

// ---------------------------------------------------------------------------
// Host orchestration (param leaf order = jax pytree flatten: sorted dict keys)
// d_in[0]=input | decoders(5x13) | downs(b,w x3) | encoders(6x13) | ending_b |
// ending_w | intro_b | intro_w | mid_dec(2x13) | mid_enc(2x13) | ups(w x3)
// Block leaves: ln_b, ln_g, [A_log, D, conv_b, conv_w, dt_proj_b, dt_proj_w,
//   in_proj_w, out_proj_w, x_proj_w], proj_b, proj_w
// ---------------------------------------------------------------------------
struct BlockP {
    const float *ln_b, *ln_g, *A_log, *D, *conv_b, *conv_w, *dt_proj_b,
                *dt_proj_w, *in_proj_w, *out_proj_w, *x_proj_w, *proj_b, *proj_w;
};

static BlockP getBlock(void* const* d_in, int base)
{
    BlockP p;
    p.ln_b       = (const float*)d_in[base + 0];
    p.ln_g       = (const float*)d_in[base + 1];
    p.A_log      = (const float*)d_in[base + 2];
    p.D          = (const float*)d_in[base + 3];
    p.conv_b     = (const float*)d_in[base + 4];
    p.conv_w     = (const float*)d_in[base + 5];
    p.dt_proj_b  = (const float*)d_in[base + 6];
    p.dt_proj_w  = (const float*)d_in[base + 7];
    p.in_proj_w  = (const float*)d_in[base + 8];
    p.out_proj_w = (const float*)d_in[base + 9];
    p.x_proj_w   = (const float*)d_in[base + 10];
    p.proj_b     = (const float*)d_in[base + 11];
    p.proj_w     = (const float*)d_in[base + 12];
    return p;
}

struct Ws {
    float *t_ln, *xz, *xcs, *dbl, *delta, *ym, *mout, *aggA, *aggB;
    _Float16 *x16, *w16;
};

static void gemm(const float* X, int lda, const float* W,
                 const float* bias, const float* resid, int ldr,
                 float* Y, int ldy, int M, int N, int K, int act,
                 const Ws& ws, hipStream_t stream)
{
    int Kp = (K + 31) & ~31;
    int Np = (N + 15) & ~15;
    {
        int tot = M * Kp;
        cvt_x_f16<<<(tot + 255) / 256, 256, 0, stream>>>(X, lda, ws.x16, M, K, Kp);
    }
    {
        int tot = Np * Kp;
        cvt_w_f16<<<(tot + 255) / 256, 256, 0, stream>>>(W, ws.w16, N, K, Kp, Np);
    }
    dim3 g((M / 16 + 3) / 4, Np / 16);
    gemm_wmma_f16<<<g, 128, 0, stream>>>(ws.x16, ws.w16, bias, resid, ldr,
                                         Y, ldy, M, N, Kp, act);
}

static void run_block(const BlockP& p, const float* x, float* dst,
                      int L, int C, const Ws& w, hipStream_t stream)
{
    const int di  = 2 * C;
    const int dtr = (C + 15) / 16;
    const int ND  = dtr + 32;

    layernorm_k<<<L, 32, 0, stream>>>(x, p.ln_g, p.ln_b, w.t_ln, L, C);

    // xz[L, 4C] = t @ in_proj_w^T
    gemm(w.t_ln, C, p.in_proj_w, nullptr, nullptr, 0, w.xz, 2 * di,
         L, 2 * di, C, 0, w, stream);

    // xcs = silu(causal_conv(xz[:, :di]) + conv_b)
    {
        int tot = L * di;
        causal_conv_silu<<<(tot + 255) / 256, 256, 0, stream>>>(
            w.xz, 2 * di, p.conv_w, p.conv_b, w.xcs, L, di);
    }

    // dbl[L, dtr+32] = xcs @ x_proj_w^T
    gemm(w.xcs, di, p.x_proj_w, nullptr, nullptr, 0, w.dbl, ND,
         L, ND, di, 0, w, stream);

    // delta = softplus(dbl[:, :dtr] @ dt_proj_w^T + dt_proj_b)
    gemm(w.dbl, ND, p.dt_proj_w, p.dt_proj_b, nullptr, 0, w.delta, di,
         L, di, dtr, 1, w, stream);

    // chunked parallel selective scan -> ym (gated)
    {
        int nch = (L + SCAN_CHUNK - 1) / SCAN_CHUNK;
        dim3 g1(nch, di / 2);
        scan_pass1<<<g1, 32, 0, stream>>>(w.delta, w.dbl + dtr, ND, w.xcs,
                                          p.A_log, w.aggA, w.aggB, L, di);
        scan_pass2<<<(di * 16 + 255) / 256, 256, 0, stream>>>(w.aggA, w.aggB,
                                                              di, nch);
        scan_pass3<<<g1, 32, 0, stream>>>(w.delta, w.dbl + dtr,
                                          w.dbl + dtr + 16, ND, w.xcs,
                                          p.A_log, p.D,
                                          w.xz + di, 2 * di,
                                          w.aggB, w.ym, L, di);
    }

    // mout[L, C] = ym @ out_proj_w^T
    gemm(w.ym, di, p.out_proj_w, nullptr, nullptr, 0, w.mout, C,
         L, C, di, 0, w, stream);

    // dst = x + mout @ proj_w^T + proj_b (residual fused in epilogue)
    gemm(w.mout, C, p.proj_w, p.proj_b, x, C, dst, C,
         L, C, C, 0, w, stream);
}

extern "C" void kernel_launch(void* const* d_in, const int* in_sizes, int n_in,
                              void* d_out, int out_size, void* d_ws, size_t ws_size,
                              hipStream_t stream)
{
    (void)in_sizes; (void)n_in; (void)out_size; (void)ws_size;
    const float* input = (const float*)d_in[0];
    float* out = (float*)d_out;

    float* f = (float*)d_ws;
    size_t off = 0;
    auto alloc = [&](size_t n) { float* p = f + off; off += (n + 7) & ~(size_t)7; return p; };
    float* skip0  = alloc(524288);   // 16384 x 32
    float* skip1  = alloc(262144);   // 4096 x 64
    float* skip2  = alloc(131072);   // 1024 x 128
    float* xlight = alloc(65536);    // 256 x 256
    float* bufA   = alloc(524288);
    float* bufB   = alloc(524288);
    Ws w;
    w.t_ln  = alloc(524288);
    w.xz    = alloc(2097152);        // 16384 x 128 max
    w.xcs   = alloc(1048576);
    w.dbl   = alloc(786432);         // 16384 x 48 max
    w.delta = alloc(1048576);
    w.ym    = alloc(1048576);
    w.mout  = alloc(524288);
    w.aggA  = alloc(65536);
    w.aggB  = alloc(65536);
    w.x16   = (_Float16*)alloc(524288);   // 1M halves (max M*Kp = 16384*64)
    w.w16   = (_Float16*)alloc(131072);   // 256K halves (max Np*Kp = 1024*256)

    const int decBase[3][3] = {{1, 14, 27}, {40, -1, -1}, {53, -1, -1}};
    const int decNum[3]     = {3, 1, 1};
    const int downBiasIdx[3] = {66, 68, 70};  // weight at +1
    const int encBase[3][3] = {{72, -1, -1}, {85, 98, -1}, {111, 124, 137}};
    const int encNum[3]     = {1, 2, 3};
    const int endingB = 150, endingW = 151, introB = 152, introW = 153;
    const int midDecBase[2] = {154, 167};
    const int midEncBase[2] = {180, 193};
    const int upsIdx[3] = {206, 207, 208};

    conv3x3_in<<<(16384 * 32 + 255) / 256, 256, 0, stream>>>(
        input, (const float*)d_in[introW], (const float*)d_in[introB],
        bufA, 128, 128);

    float* cur = bufA;
    float* skips[3] = {skip0, skip1, skip2};
    int H = 128, C = 32;

    for (int s = 0; s < 3; ++s) {
        int L = H * H;
        for (int bI = 0; bI < encNum[s]; ++bI) {
            BlockP bp = getBlock(d_in, encBase[s][bI]);
            float* dst = (bI == encNum[s] - 1) ? skips[s]
                                               : ((cur == bufA) ? bufB : bufA);
            run_block(bp, cur, dst, L, C, w, stream);
            cur = dst;
        }
        {
            int OC = 2 * C, Ho = H / 2;
            int tot = Ho * Ho * OC;
            conv2x2_down<<<(tot + 255) / 256, 256, 0, stream>>>(
                cur, (const float*)d_in[downBiasIdx[s] + 1],
                (const float*)d_in[downBiasIdx[s]], bufA, H, H, C);
            cur = bufA;
        }
        H /= 2; C *= 2;
    }

    {
        int L = H * H;   // 256 tokens, C = 256
        for (int bI = 0; bI < 2; ++bI) {
            BlockP bp = getBlock(d_in, midEncBase[bI]);
            float* dst = (cur == bufA) ? bufB : bufA;
            run_block(bp, cur, dst, L, C, w, stream);
            cur = dst;
        }
        (void)hipMemcpyAsync(xlight, cur, (size_t)L * C * sizeof(float),
                             hipMemcpyDeviceToDevice, stream);
        for (int bI = 0; bI < 2; ++bI) {
            BlockP bp = getBlock(d_in, midDecBase[bI]);
            float* dst = (cur == bufA) ? bufB : bufA;
            run_block(bp, cur, dst, L, C, w, stream);
            cur = dst;
        }
        add_inplace<<<(L * C + 255) / 256, 256, 0, stream>>>(cur, xlight, L * C);
    }

    float* decSkips[3] = {skip2, skip1, skip0};
    for (int s = 0; s < 3; ++s) {
        int L = H * H;
        // 1x1 up conv via WMMA GEMM: [L, C] @ up_w[2C, C]^T -> xz[L, 2C]
        gemm(cur, C, (const float*)d_in[upsIdx[s]], nullptr, nullptr, 0,
             w.xz, 2 * C, L, 2 * C, C, 0, w, stream);
        int Co = C / 2;
        {
            int tot = 4 * L * Co;
            pixshuf_add<<<(tot + 255) / 256, 256, 0, stream>>>(
                w.xz, decSkips[s], bufA, H, H, Co);
            cur = bufA;
        }
        H *= 2; C = Co;
        int Ln = H * H;
        for (int bI = 0; bI < decNum[s]; ++bI) {
            BlockP bp = getBlock(d_in, decBase[s][bI]);
            float* dst = (cur == bufA) ? bufB : bufA;
            run_block(bp, cur, dst, Ln, C, w, stream);
            cur = dst;
        }
    }

    conv3x3_out<<<(3 * 128 * 128 + 255) / 256, 256, 0, stream>>>(
        cur, (const float*)d_in[endingW], (const float*)d_in[endingB],
        input, out, 128, 128);
}
